// DirectVoxGO_39462159516244
// MI455X (gfx1250) — compile-verified, matrix operand order checked by reading, not run on hardware
//
#include <hip/hip_runtime.h>
#include <hip/hip_bf16.h>
#include <math.h>

// ---------------------------------------------------------------------------
// DirectVoxGO forward on MI455X (gfx1250, wave32, WMMA f16)
// ---------------------------------------------------------------------------
#define RESG      160
#define RES3      4096000            // 160^3
#define XS        25600              // x stride
#define YS        160                // y stride
#define NRAYS     8192
#define NSAMP     160
#define NTOT      (NRAYS * NSAMP)    // 1,310,720
#define ACT_SHIFT (-13.81551056f)

typedef _Float16 v8h  __attribute__((ext_vector_type(8)));
typedef _Float16 v16h __attribute__((ext_vector_type(16)));
typedef float    v8f  __attribute__((ext_vector_type(8)));

// load a WMMA f16 A/B operand fragment: halves 0-7 at p, halves 8-15 at p+16
__device__ __forceinline__ v16h load_ab(const _Float16* p) {
    v8h lo = *(const v8h*)p;
    v8h hi = *(const v8h*)(p + 16);
    return __builtin_shufflevector(lo, hi, 0,1,2,3,4,5,6,7,8,9,10,11,12,13,14,15);
}

__device__ __forceinline__ void mkInterp(float x, float y, float z,
                                         int& i000, float* cw) {
    float px = fminf(fmaxf((x + 1.0f) * 0.5f * 159.0f, 0.0f), 159.0f);
    float py = fminf(fmaxf((y + 1.0f) * 0.5f * 159.0f, 0.0f), 159.0f);
    float pz = fminf(fmaxf((z + 1.0f) * 0.5f * 159.0f, 0.0f), 159.0f);
    int ix = min((int)px, 158), iy = min((int)py, 158), iz = min((int)pz, 158);
    float tx = px - (float)ix, ty = py - (float)iy, tz = pz - (float)iz;
    i000 = ix * XS + iy * YS + iz;
    float ux = 1.0f - tx, uy = 1.0f - ty, uz = 1.0f - tz;
    cw[0] = ux*uy*uz;  cw[1] = tx*uy*uz;  cw[2] = ux*ty*uz;  cw[3] = tx*ty*uz;
    cw[4] = ux*uy*tz;  cw[5] = tx*uy*tz;  cw[6] = ux*ty*tz;  cw[7] = tx*ty*tz;
}

__device__ __forceinline__ float interp1(const float* __restrict__ g, int i000,
                                         const float* cw) {
    return cw[0]*g[i000]        + cw[1]*g[i000+XS]
         + cw[2]*g[i000+YS]     + cw[3]*g[i000+XS+YS]
         + cw[4]*g[i000+1]      + cw[5]*g[i000+XS+1]
         + cw[6]*g[i000+YS+1]   + cw[7]*g[i000+XS+YS+1];
}

// ---------------------------------------------------------------------------
// Kernel 0: transpose + f16-convert MLP weights (pad K: 72->96, N: 3->16)
//   w0T: [128][96]  w1T: [128][128]  w2T: [16][128]
// ---------------------------------------------------------------------------
__global__ void prep_weights(const float* __restrict__ w0,
                             const float* __restrict__ w1,
                             const float* __restrict__ w2,
                             _Float16* __restrict__ w0T,
                             _Float16* __restrict__ w1T,
                             _Float16* __restrict__ w2T) {
    int i = blockIdx.x * blockDim.x + threadIdx.x;
    if (i < 128 * 96) {
        int n = i / 96, k = i % 96;
        w0T[i] = (_Float16)(k < 72 ? w0[k * 128 + n] : 0.0f);
    } else if (i < 128 * 96 + 128 * 128) {
        int j = i - 128 * 96;
        int n = j / 128, k = j % 128;
        w1T[j] = (_Float16)w1[k * 128 + n];
    } else if (i < 128 * 96 + 128 * 128 + 16 * 128) {
        int j = i - 128 * 96 - 128 * 128;
        int n = j / 128, k = j % 128;
        w2T[j] = (_Float16)(n < 3 ? w2[k * 3 + n] : 0.0f);
    }
}

// ---------------------------------------------------------------------------
// Kernel 1: per-sample density interpolation
// ---------------------------------------------------------------------------
__global__ void dens_kernel(const float* __restrict__ ro,
                            const float* __restrict__ rd,
                            const float* __restrict__ grid,
                            float* __restrict__ dens) {
    int idx = blockIdx.x * blockDim.x + threadIdx.x;   // exact: NTOT
    int r = idx / NSAMP, s = idx % NSAMP;
    float t = 0.05f + (0.65f / 159.0f) * (float)s;
    float x = ro[r*3+0] + rd[r*3+0] * t;
    float y = ro[r*3+1] + rd[r*3+1] * t;
    float z = ro[r*3+2] + rd[r*3+2] * t;
    int i000; float cw[8];
    mkInterp(x, y, z, i000, cw);
    dens[idx] = interp1(grid, i000, cw);
}

// ---------------------------------------------------------------------------
// Kernel 2: per-ray transmittance scan -> weights[R,160], Tlast[R]
// ---------------------------------------------------------------------------
__global__ void scan_kernel(const float* __restrict__ dens,
                            float* __restrict__ wts,
                            float* __restrict__ Tlast) {
    int r = blockIdx.x * blockDim.x + threadIdx.x;     // exact: NRAYS
    float T = 1.0f;
    for (int s = 0; s < NSAMP; ++s) {
        float xx = dens[r * NSAMP + s] + ACT_SHIFT;
        float sp = (xx > 20.0f) ? xx : log1pf(expf(xx));   // softplus
        float alpha = 1.0f - expf(-sp);
        wts[r * NSAMP + s] = alpha * T;
        T *= fmaxf(1.0f - alpha, 1e-10f);
    }
    Tlast[r] = T;
}

// ---------------------------------------------------------------------------
// Kernel 3: fused featurize + 3-layer WMMA MLP + per-tile composite partials
//   one wave (32 lanes) per 16-sample tile; 4 waves / workgroup
// ---------------------------------------------------------------------------
__global__ void __launch_bounds__(128)
mlp_kernel(const float* __restrict__ ro, const float* __restrict__ rd,
           const float* __restrict__ vd, const float* __restrict__ k0,
           const _Float16* __restrict__ w0T, const _Float16* __restrict__ w1T,
           const _Float16* __restrict__ w2T,
           const float* __restrict__ b0, const float* __restrict__ b1,
           const float* __restrict__ b2,
           const float* __restrict__ wts, float* __restrict__ partial) {
    __shared__ __align__(16) _Float16 featS[4][16][96];   // 12 KB
    __shared__ __align__(16) _Float16 h0S[4][16][128];    // 16 KB
    __shared__ __align__(16) _Float16 h1S[4][16][128];    // 16 KB

    const int w    = threadIdx.x >> 5;     // wave in WG
    const int lane = threadIdx.x & 31;
    const int nl   = lane & 15;            // N / M-row index within tile
    const int half = lane >> 4;            // K-half selector
    const int tile = blockIdx.x * 4 + w;   // 0 .. 81919
    const int ray  = tile / 10;
    const int s0   = (tile % 10) * 16;

    // ---------------- feature phase: feat[16][96] f16 into LDS -------------
    {
        const int sl = nl;                 // local sample
        const int s  = s0 + sl;
        float t = 0.05f + (0.65f / 159.0f) * (float)s;
        float x = ro[ray*3+0] + rd[ray*3+0] * t;
        float y = ro[ray*3+1] + rd[ray*3+1] * t;
        float z = ro[ray*3+2] + rd[ray*3+2] * t;
        int i000; float cw[8];
        mkInterp(x, y, z, i000, cw);
        // k0 channels split across the two half-waves (6 each)
        int cbeg = half * 6;
        for (int c = cbeg; c < cbeg + 6; ++c)
            featS[w][sl][c] =
                (_Float16)interp1(k0 + (size_t)c * RES3, i000, cw);
        if (half == 0) {
            // xyz positional encoding: feat[12..44]
            featS[w][sl][12] = (_Float16)x;
            featS[w][sl][13] = (_Float16)y;
            featS[w][sl][14] = (_Float16)z;
            float p[3] = {x, y, z};
            for (int d = 0; d < 3; ++d) {
                float f = 1.0f;
                for (int q = 0; q < 5; ++q) {
                    float a = p[d] * f;
                    featS[w][sl][15 + d*5 + q] = (_Float16)sinf(a);
                    featS[w][sl][30 + d*5 + q] = (_Float16)cosf(a);
                    f *= 2.0f;
                }
            }
        } else {
            // view positional encoding: feat[45..71], pad feat[72..95]=0
            float vx = vd[ray*3+0], vy = vd[ray*3+1], vz = vd[ray*3+2];
            featS[w][sl][45] = (_Float16)vx;
            featS[w][sl][46] = (_Float16)vy;
            featS[w][sl][47] = (_Float16)vz;
            float p[3] = {vx, vy, vz};
            for (int d = 0; d < 3; ++d) {
                float f = 1.0f;
                for (int q = 0; q < 4; ++q) {
                    float a = p[d] * f;
                    featS[w][sl][48 + d*4 + q] = (_Float16)sinf(a);
                    featS[w][sl][60 + d*4 + q] = (_Float16)cosf(a);
                    f *= 2.0f;
                }
            }
            for (int kpad = 72; kpad < 96; ++kpad)
                featS[w][sl][kpad] = (_Float16)0.0f;
        }
    }
    __syncthreads();

    const int rowA = nl;
    const int kofs = half * 8;             // K-base per A/B layout

    // ---------------- layer 0: [16,96] x [96,128] -> h0 --------------------
    {
        v16h a[3];
#pragma unroll
        for (int kk = 0; kk < 3; ++kk)
            a[kk] = load_ab(&featS[w][rowA][kk * 32 + kofs]);
#pragma unroll
        for (int j = 0; j < 8; ++j) {
            v8f acc = {};
#pragma unroll
            for (int kk = 0; kk < 3; ++kk) {
                v16h b = load_ab(&w0T[(j*16 + nl) * 96 + kk * 32 + kofs]);
                acc = __builtin_amdgcn_wmma_f32_16x16x32_f16(
                          false, a[kk], false, b, (short)0, acc, false, false);
            }
            float bias = b0[j * 16 + nl];
#pragma unroll
            for (int v = 0; v < 8; ++v)
                h0S[w][half*8 + v][j*16 + nl] =
                    (_Float16)fmaxf(acc[v] + bias, 0.0f);
        }
    }
    __syncthreads();

    // ---------------- layer 1: [16,128] x [128,128] -> h1 ------------------
    {
        v16h a[4];
#pragma unroll
        for (int kk = 0; kk < 4; ++kk)
            a[kk] = load_ab(&h0S[w][rowA][kk * 32 + kofs]);
#pragma unroll
        for (int j = 0; j < 8; ++j) {
            v8f acc = {};
#pragma unroll
            for (int kk = 0; kk < 4; ++kk) {
                v16h b = load_ab(&w1T[(j*16 + nl) * 128 + kk * 32 + kofs]);
                acc = __builtin_amdgcn_wmma_f32_16x16x32_f16(
                          false, a[kk], false, b, (short)0, acc, false, false);
            }
            float bias = b1[j * 16 + nl];
#pragma unroll
            for (int v = 0; v < 8; ++v)
                h1S[w][half*8 + v][j*16 + nl] =
                    (_Float16)fmaxf(acc[v] + bias, 0.0f);
        }
    }
    __syncthreads();

    // ---------------- layer 2: [16,128] x [128,16(3)] + composite ----------
    {
        v16h a[4];
#pragma unroll
        for (int kk = 0; kk < 4; ++kk)
            a[kk] = load_ab(&h1S[w][rowA][kk * 32 + kofs]);
        v8f acc = {};
#pragma unroll
        for (int kk = 0; kk < 4; ++kk) {
            v16h b = load_ab(&w2T[nl * 128 + kk * 32 + kofs]);
            acc = __builtin_amdgcn_wmma_f32_16x16x32_f16(
                      false, a[kk], false, b, (short)0, acc, false, false);
        }
        float bias = (nl < 3) ? b2[nl] : 0.0f;
        float sum = 0.0f;
#pragma unroll
        for (int v = 0; v < 8; ++v) {
            int m = half * 8 + v;
            float sig = 1.0f / (1.0f + expf(-(acc[v] + bias)));
            sum += sig * wts[ray * NSAMP + s0 + m];
        }
        float other = __shfl_xor(sum, 16, 32);   // combine the two M-halves
        if (nl < 3 && half == 0)
            partial[tile * 3 + nl] = sum + other;
    }
}

// ---------------------------------------------------------------------------
// Kernel 4: reduce 10 tile partials per ray + background term
// ---------------------------------------------------------------------------
__global__ void reduce_kernel(const float* __restrict__ partial,
                              const float* __restrict__ Tlast,
                              float* __restrict__ out) {
    int i = blockIdx.x * blockDim.x + threadIdx.x;     // exact: 24576
    int r = i / 3, c = i % 3;
    float acc = Tlast[r];
    for (int t = 0; t < 10; ++t)
        acc += partial[(r * 10 + t) * 3 + c];
    out[i] = acc;
}

// ---------------------------------------------------------------------------
extern "C" void kernel_launch(void* const* d_in, const int* in_sizes, int n_in,
                              void* d_out, int out_size, void* d_ws, size_t ws_size,
                              hipStream_t stream) {
    const float* ro  = (const float*)d_in[0];
    const float* rd  = (const float*)d_in[1];
    const float* vd  = (const float*)d_in[2];
    const float* den = (const float*)d_in[3];
    const float* k0  = (const float*)d_in[4];
    const float* w0  = (const float*)d_in[5];
    const float* b0  = (const float*)d_in[6];
    const float* w1  = (const float*)d_in[7];
    const float* b1  = (const float*)d_in[8];
    const float* w2  = (const float*)d_in[9];
    const float* b2  = (const float*)d_in[10];
    float* out = (float*)d_out;

    char* ws = (char*)d_ws;
    float*    dens    = (float*)   (ws);                       //  5,242,880 B
    float*    wts     = (float*)   (ws + 5242880);             //  5,242,880 B
    _Float16* w0T     = (_Float16*)(ws + 10485760);            //     24,576 B
    _Float16* w1T     = (_Float16*)(ws + 10510336);            //     32,768 B
    _Float16* w2T     = (_Float16*)(ws + 10543104);            //      4,096 B
    float*    Tlast   = (float*)   (ws + 10547200);            //     32,768 B
    float*    partial = (float*)   (ws + 10579968);            //    983,040 B

    prep_weights<<<120, 256, 0, stream>>>(w0, w1, w2, w0T, w1T, w2T);
    dens_kernel<<<NTOT / 256, 256, 0, stream>>>(ro, rd, den, dens);
    scan_kernel<<<NRAYS / 256, 256, 0, stream>>>(dens, wts, Tlast);
    mlp_kernel<<<20480, 128, 0, stream>>>(ro, rd, vd, k0, w0T, w1T, w2T,
                                          b0, b1, b2, wts, partial);
    reduce_kernel<<<96, 256, 0, stream>>>(partial, Tlast, out);
}
